// MoDESSkippedQwen3MoeSparseMoeBlock_34119220199815
// MI455X (gfx1250) — matrix operand order repaired and checked by reading, not compiled, not run on hardware
//
#include <hip/hip_runtime.h>
#include <hip/hip_bf16.h>

// ---------------------------------------------------------------------------
// MoE sparse block (Qwen3-style, tau-gated top-8) for gfx1250 / MI455X.
// fp32 in/out; matmuls run on v_wmma_f32_16x16x32_bf16 (x converted once,
// weights converted fp32->bf16 in-register while streaming from HBM).
// Each block covers 32 tokens (2 M-subtiles) so every B fragment loaded from
// the fp32 weights feeds two WMMAs -> 2x arithmetic intensity vs 16-row tiles.
// ---------------------------------------------------------------------------

#define NUM_EXPERTS 32
#define HIDDEN      2048
#define INTER       768
#define TWOI        1536
#define TOKENS      2048           // BATCH*SEQ = 2*1024
#define TOPK        8
#define TAU         0.05f
#define ROWS        32             // tokens per block (2 x 16-row M-subtiles)
#define MAX_TILES   (TOKENS / ROWS)
#define HPAD        776            // LDS row stride for h (bank-spread pad)

typedef __attribute__((ext_vector_type(16))) __bf16 v16bf;
typedef __attribute__((ext_vector_type(8)))  float  v8f;
typedef unsigned int u32;

// fast reciprocal: single v_rcp_f32
static __device__ __forceinline__ float frcp(float x) {
    return __builtin_amdgcn_rcpf(x);
}

// Load an A-fragment (16x32 bf16, MxK) for this lane.
// Per ISA layout: lane holds 8 bf16 at K = base0..base0+7 (VGPR0-3) and
// 8 bf16 at K = base0+16..base0+23 (VGPR4-7), base0 = (lane>=16)*8.
// p points at (row_base + k0 + base0).
static __device__ __forceinline__ v16bf load_a16(const __bf16* p) {
    union { uint4 q[2]; v16bf v; } r;
    r.q[0] = *(const uint4*)(p);
    r.q[1] = *(const uint4*)(p + 16);
    return r.v;
}

// Load a B-fragment (32x16 bf16, KxN) from fp32 memory with conversion.
// Lane: n = lane&15, kbase = (lane>>4)*8 -> 8 fp32 at W[n][k0 + kbase .. +7]
// and 8 at W[n][k0 + kbase + 16 .. +23]. Conversion via HW packed bf16 cvt (RNE).
static __device__ __forceinline__ v16bf load_b_f32(const float* p) {
    float4 w0 = ((const float4*)p)[0];
    float4 w1 = ((const float4*)p)[1];
    float4 w2 = ((const float4*)p)[4];
    float4 w3 = ((const float4*)p)[5];
    union { __bf16 b[16]; v16bf v; } r;
    r.b[0]  = (__bf16)w0.x; r.b[1]  = (__bf16)w0.y; r.b[2]  = (__bf16)w0.z; r.b[3]  = (__bf16)w0.w;
    r.b[4]  = (__bf16)w1.x; r.b[5]  = (__bf16)w1.y; r.b[6]  = (__bf16)w1.z; r.b[7]  = (__bf16)w1.w;
    r.b[8]  = (__bf16)w2.x; r.b[9]  = (__bf16)w2.y; r.b[10] = (__bf16)w2.z; r.b[11] = (__bf16)w2.w;
    r.b[12] = (__bf16)w3.x; r.b[13] = (__bf16)w3.y; r.b[14] = (__bf16)w3.z; r.b[15] = (__bf16)w3.w;
    return r.v;
}

#define WMMA_BF16(A, B, C) \
    __builtin_amdgcn_wmma_f32_16x16x32_bf16(false, (A), false, (B), (short)0, (C), false, false)

// ---------------------------------------------------------------------------
// Router: one wave32 per token; lane e owns expert e (E == wave width).
// ---------------------------------------------------------------------------
__global__ __launch_bounds__(32)
void router_kernel(const float* __restrict__ x, const float* __restrict__ gw,
                   const float* __restrict__ alpha_p,
                   float* __restrict__ combine, int* __restrict__ counts,
                   int* __restrict__ toks) {
    const int t = blockIdx.x;
    const int e = threadIdx.x;

    const float4* xr = (const float4*)(x + (size_t)t * HIDDEN);
    const float4* gr = (const float4*)(gw + (size_t)e * HIDDEN);
    float acc = 0.0f;
    #pragma unroll 4
    for (int i = 0; i < HIDDEN / 4; ++i) {
        float4 a = xr[i], b = gr[i];
        acc += a.x * b.x + a.y * b.y + a.z * b.z + a.w * b.w;
    }

    // softmax across the wave
    float m = acc;
    for (int off = 16; off; off >>= 1) m = fmaxf(m, __shfl_xor(m, off, 32));
    float ex = __expf(acc - m);
    float s = ex;
    for (int off = 16; off; off >>= 1) s += __shfl_xor(s, off, 32);
    float prob = ex * frcp(s);

    // top-8 by iterated wave argmax (ties -> lowest expert index)
    bool sel = false;
    for (int it = 0; it < TOPK; ++it) {
        float v = sel ? -1.0f : prob;
        int idx = e;
        for (int off = 16; off; off >>= 1) {
            float ov = __shfl_xor(v, off, 32);
            int   oi = __shfl_xor(idx, off, 32);
            if (ov > v || (ov == v && oi < idx)) { v = ov; idx = oi; }
        }
        if (e == idx) sel = true;
    }

    float tsum = sel ? prob : 0.0f;
    for (int off = 16; off; off >>= 1) tsum += __shfl_xor(tsum, off, 32);
    tsum = fmaxf(tsum, 1e-12f);
    float topv = prob * frcp(tsum);

    const float alpha = alpha_p[0];
    float scaled = topv * alpha;
    bool active = sel && (scaled >= TAU);

    if (__ballot(active) == 0ull) {       // empty-row fallback: argmax(scaled)
        float v = sel ? scaled : -3.4e38f;
        int idx = e;
        for (int off = 16; off; off >>= 1) {
            float ov = __shfl_xor(v, off, 32);
            int   oi = __shfl_xor(idx, off, 32);
            if (ov > v || (ov == v && oi < idx)) { v = ov; idx = oi; }
        }
        active = (e == idx);
    }

    float rsum = active ? topv : 0.0f;
    for (int off = 16; off; off >>= 1) rsum += __shfl_xor(rsum, off, 32);
    rsum = fmaxf(rsum, 1e-12f);
    float wgt = active ? (topv * frcp(rsum)) : 0.0f;

    combine[(size_t)t * NUM_EXPERTS + e] = wgt;
    if (wgt > 0.0f) {
        int pos = atomicAdd(&counts[e], 1);
        toks[e * TOKENS + pos] = t;
    }
}

// ---------------------------------------------------------------------------
// x fp32 -> bf16 (packed b128 stores)
// ---------------------------------------------------------------------------
__global__ __launch_bounds__(256)
void cvt_kernel(const float* __restrict__ x, __bf16* __restrict__ xb) {
    size_t i = ((size_t)blockIdx.x * 256 + threadIdx.x) * 8;
    float4 a = *(const float4*)(x + i);
    float4 b = *(const float4*)(x + i + 4);
    union { __bf16 s[8]; uint4 q; } r;
    r.s[0] = (__bf16)a.x; r.s[1] = (__bf16)a.y; r.s[2] = (__bf16)a.z; r.s[3] = (__bf16)a.w;
    r.s[4] = (__bf16)b.x; r.s[5] = (__bf16)b.y; r.s[6] = (__bf16)b.z; r.s[7] = (__bf16)b.w;
    *(uint4*)(xb + i) = r.q;
}

// ---------------------------------------------------------------------------
// Fused expert kernel: one block per (expert, 32-token tile of its list).
// Phase 1: gu = x_tile @ Wgu^T via WMMA bf16 (each B fragment feeds both
//          M-subtiles), SwiGLU in-register -> h (32x768 bf16) in LDS.
// Phase 2: out += (h @ Wd^T) * route_weight via WMMA bf16 + f32 atomics.
// ---------------------------------------------------------------------------
__global__ __launch_bounds__(256)
void expert_kernel(const __bf16* __restrict__ xbf,
                   const float* __restrict__ wgu,
                   const float* __restrict__ wd,
                   const float* __restrict__ combine,
                   const int* __restrict__ counts,
                   const int* __restrict__ toks,
                   float* __restrict__ out) {
    const int e    = blockIdx.x / MAX_TILES;
    const int tile = blockIdx.x % MAX_TILES;
    const int cnt  = counts[e];
    const int r0   = tile * ROWS;
    if (r0 >= cnt) return;                // block-uniform exit: EXEC stays full

    __shared__ int   s_tok[ROWS];
    __shared__ float s_w[ROWS];
    __shared__ __align__(16) __bf16 s_h[ROWS][HPAD];

    const int tid  = threadIdx.x;
    const int lane = tid & 31;
    const int wv   = tid >> 5;
    const int hm   = lane & 15;           // M row / N col within 16x16 tile
    const int hi   = lane >> 4;           // half-wave select

    if (tid < ROWS) {
        int r  = r0 + tid;
        int rc = (r < cnt) ? r : (cnt - 1);
        int tk = toks[e * TOKENS + rc];
        s_tok[tid] = tk;
        s_w[tid]   = (r < cnt) ? combine[(size_t)tk * NUM_EXPERTS + e] : 0.0f;
    }
    __syncthreads();

    const __bf16* arow0 = xbf + (size_t)s_tok[hm]      * HIDDEN + hi * 8;
    const __bf16* arow1 = xbf + (size_t)s_tok[16 + hm] * HIDDEN + hi * 8;
    const float*  wgu_e = wgu + (size_t)e * TWOI * HIDDEN;
    const float*  wd_e  = wd  + (size_t)e * HIDDEN * INTER;

    // ------- Phase 1: paired (g, u) tiles x 2 M-subtiles + SwiGLU ---------
    for (int pt = wv; pt < INTER / 16; pt += 8) {
        v8f accg0 = {}, accu0 = {};
        v8f accg1 = {}, accu1 = {};
        const int ng = pt * 16 + hm;
        const float* bg = wgu_e + (size_t)ng * HIDDEN + hi * 8;
        const float* bu = bg + (size_t)INTER * HIDDEN;      // u rows at +768
        #pragma unroll 2
        for (int k0 = 0; k0 < HIDDEN; k0 += 32) {
            v16bf A0 = load_a16(arow0 + k0);
            v16bf A1 = load_a16(arow1 + k0);
            v16bf Bg = load_b_f32(bg + k0);
            v16bf Bu = load_b_f32(bu + k0);
            __builtin_prefetch(bg + k0 + 128, 0, 0);
            __builtin_prefetch(bu + k0 + 128, 0, 0);
            accg0 = WMMA_BF16(A0, Bg, accg0);
            accg1 = WMMA_BF16(A1, Bg, accg1);
            accu0 = WMMA_BF16(A0, Bu, accu0);
            accu1 = WMMA_BF16(A1, Bu, accu1);
        }
        const int rbase = hi * 8;
        const int col   = pt * 16 + hm;
        #pragma unroll
        for (int j = 0; j < 8; ++j) {
            float g0 = accg0[j], u0 = accu0[j];
            float g1 = accg1[j], u1 = accu1[j];
            s_h[j + rbase][col]      = (__bf16)(g0 * frcp(1.0f + __expf(-g0)) * u0);
            s_h[16 + j + rbase][col] = (__bf16)(g1 * frcp(1.0f + __expf(-g1)) * u1);
        }
    }
    __syncthreads();

    // ------- Phase 2: h @ Wd^T (B shared by both M-subtiles), scatter ------
    const __bf16* ap0 = &s_h[hm][hi * 8];
    const __bf16* ap1 = &s_h[16 + hm][hi * 8];
    for (int nt = wv; nt < HIDDEN / 16; nt += 8) {
        v8f acc0 = {}, acc1 = {};
        const int n = nt * 16 + hm;
        const float* bp = wd_e + (size_t)n * INTER + hi * 8;
        #pragma unroll 2
        for (int k0 = 0; k0 < INTER; k0 += 32) {
            v16bf A0 = load_a16(ap0 + k0);                  // ds_read_b128 x2
            v16bf A1 = load_a16(ap1 + k0);
            v16bf B  = load_b_f32(bp + k0);
            __builtin_prefetch(bp + k0 + 128, 0, 0);
            acc0 = WMMA_BF16(A0, B, acc0);
            acc1 = WMMA_BF16(A1, B, acc1);
        }
        const int rbase = hi * 8;
        const int col   = nt * 16 + hm;
        #pragma unroll
        for (int j = 0; j < 8; ++j) {
            int row0 = j + rbase;
            int row1 = 16 + j + rbase;
            atomicAdd(&out[(size_t)s_tok[row0] * HIDDEN + col], acc0[j] * s_w[row0]);
            atomicAdd(&out[(size_t)s_tok[row1] * HIDDEN + col], acc1[j] * s_w[row1]);
        }
    }
}

// ---------------------------------------------------------------------------
// Host-side launcher (graph-capture safe: only async ops on `stream`)
// ---------------------------------------------------------------------------
extern "C" void kernel_launch(void* const* d_in, const int* in_sizes, int n_in,
                              void* d_out, int out_size, void* d_ws, size_t ws_size,
                              hipStream_t stream) {
    const float* x     = (const float*)d_in[0];   // [T, H]
    const float* gw    = (const float*)d_in[1];   // [E, H]
    const float* wgu   = (const float*)d_in[2];   // [E, 2I, H]
    const float* wd    = (const float*)d_in[3];   // [E, H, I]
    const float* alpha = (const float*)d_in[4];   // scalar
    float* out = (float*)d_out;

    // workspace layout
    char* ws = (char*)d_ws;
    float* combine = (float*)ws;                                  // T*E f32
    size_t off = (size_t)TOKENS * NUM_EXPERTS * sizeof(float);
    int* counts = (int*)(ws + off);                               // E i32
    off += 1024;
    int* toks = (int*)(ws + off);                                 // E*T i32
    off += (size_t)NUM_EXPERTS * TOKENS * sizeof(int);
    __bf16* xbf = (__bf16*)(ws + off);                            // T*H bf16

    hipMemsetAsync(counts, 0, NUM_EXPERTS * sizeof(int), stream);
    hipMemsetAsync(out, 0, (size_t)TOKENS * HIDDEN * sizeof(float), stream);

    router_kernel<<<TOKENS, 32, 0, stream>>>(x, gw, alpha, combine, counts, toks);
    cvt_kernel<<<(TOKENS * HIDDEN) / (256 * 8), 256, 0, stream>>>(x, xbf);
    expert_kernel<<<NUM_EXPERTS * MAX_TILES, 256, 0, stream>>>(
        xbf, wgu, wd, combine, counts, toks, out);
}
